// RFS_76218489635120
// MI455X (gfx1250) — matrix operand order repaired and checked by reading, not compiled, not run on hardware
//
#include <hip/hip_runtime.h>
#include <hip/hip_bf16.h>

typedef __attribute__((ext_vector_type(16))) _Float16 v16h;
typedef __attribute__((ext_vector_type(8)))  float    v8f;

#define BSZ 1024

// ---------------------------------------------------------------------------
// Conv + bias + ReLU + BN-stats, implicit GEMM with v_wmma_f32_16x16x32_f16.
// M = 16 output pixels (flattened b*HOUT*WOUT tile), N = 24 ch (2x16 tiles),
// K = CIN*9 padded to 32. One wave per M-tile, 8 waves / 256-thread block.
// Weights staged to LDS via ASYNC global->LDS copies (ASYNCcnt); padded
// region zeroed with plain DS stores (disjoint addresses, no ordering hazard).
// ---------------------------------------------------------------------------
template<typename TI, int CIN, int HIN, int WIN, int HOUT, int WOUT>
__global__ __launch_bounds__(256, 4) void conv_wmma_kernel(
    const TI*     __restrict__ x,       // [B,CIN,HIN,WIN]
    const float*  __restrict__ w,       // [24,CIN,3,3]
    const float*  __restrict__ bias,    // [24]
    const float*  __restrict__ aff_in,  // [48] scale|shift for input, or null
    _Float16*     __restrict__ y,       // [B,24,HOUT,WOUT]
    float*        __restrict__ stats)   // [48] sum|sumsq accumulators
{
  constexpr int K    = CIN * 9;
  constexpr int KP   = ((K + 31) / 32) * 32;
  constexpr int KC   = KP / 32;
  constexpr int NTOT = BSZ * HOUT * WOUT;
  constexpr int TILES = (NTOT + 15) / 16;

  __shared__ float wlds[32 * KP];      // f32 weights, layout [cout(32)][KP]

  // Stage weights: valid elements via async DMA global->LDS, pad with zeros.
  for (int i = threadIdx.x; i < 32 * KP; i += 256) {
    int co = i / KP, k = i % KP;
    if (co < 24 && k < K) {
      auto lptr = (__attribute__((address_space(3))) float*)&wlds[i];
      unsigned loff = (unsigned)(uintptr_t)lptr;
      const float* gp = &w[co * K + k];
      asm volatile("global_load_async_to_lds_b32 %0, %1, off"
                   :: "v"(loff), "v"(gp) : "memory");
    } else {
      wlds[i] = 0.0f;
    }
  }
  asm volatile("s_wait_asynccnt 0x0" ::: "memory");
  __syncthreads();

  const int wave = threadIdx.x >> 5;
  const int lane = threadIdx.x & 31;
  const int tile = blockIdx.x * 8 + wave;
  if (tile >= TILES) return;           // wave-uniform: EXEC stays all-ones

  // A-fragment row for this lane (16-bit A 16x32 layout: lane&15 = row M).
  const int m  = lane & 15;
  const int p  = tile * 16 + m;
  const bool pv = p < NTOT;
  const int pp = pv ? p : 0;
  const int b  = pp / (HOUT * WOUT);
  const int rr = pp % (HOUT * WOUT);
  const int oh = rr / WOUT;
  const int ow = rr % WOUT;

  // K index for half h of the A fragment (per ISA 16-bit A 16x32 table).
  int kmap[16];
#pragma unroll
  for (int h = 0; h < 16; ++h) {
    int j = h >> 1;
    int base = (j < 4) ? (2 * j) : (2 * j + 8);
    kmap[h] = ((lane >> 4) << 3) + base + (h & 1);
  }

  v8f acc0 = {};
  v8f acc1 = {};

#pragma unroll 1
  for (int kc = 0; kc < KC; ++kc) {
    // Prefetch a representative element of the next K-chunk (speculative).
    if (kc + 1 < KC) {
      int kg = (kc + 1) * 32 + kmap[0];
      if (pv && kg < K) {
        int cin = kg / 9, rem = kg % 9;
        int ky = rem / 3, kx = rem % 3;
        int ih = oh * 2 - 1 + ky, iw = ow * 2 - 1 + kx;
        if (ih >= 0 && ih < HIN && iw >= 0 && iw < WIN)
          __builtin_prefetch(
              &x[((size_t)(b * CIN + cin)) * (HIN * WIN) + ih * WIN + iw], 0, 1);
      }
    }

    // ---- A fragment: im2col gather (row m, 16 K values) ----
    v16h a;
#pragma unroll
    for (int h = 0; h < 16; ++h) {
      int kg = kc * 32 + kmap[h];
      float v = 0.0f;
      if (pv && kg < K) {
        int cin = kg / 9, rem = kg % 9;
        int ky = rem / 3, kx = rem % 3;
        int ih = oh * 2 - 1 + ky;
        int iw = ow * 2 - 1 + kx;
        if (ih >= 0 && ih < HIN && iw >= 0 && iw < WIN) {
          v = (float)x[((size_t)(b * CIN + cin)) * (HIN * WIN) + ih * WIN + iw];
          if (aff_in) v = v * aff_in[cin] + aff_in[24 + cin];
        }
      }
      a[h] = (_Float16)v;
    }

    // ---- B fragments from LDS weights (col = lane&15, K split by lane>>4) ----
    const int n  = lane & 15;
    const int kb = kc * 32 + ((lane >> 4) << 3);
    v16h b0, b1;
#pragma unroll
    for (int h = 0; h < 16; ++h) {
      b0[h] = (_Float16)wlds[n * KP + kb + h + ((h >> 3) << 3)];
      b1[h] = (_Float16)wlds[(16 + n) * KP + kb + h + ((h >> 3) << 3)];
    }

    acc0 = __builtin_amdgcn_wmma_f32_16x16x32_f16(false, a, false, b0, (short)0,
                                                  acc0, false, false);
    acc1 = __builtin_amdgcn_wmma_f32_16x16x32_f16(false, a, false, b1, (short)0,
                                                  acc1, false, false);
  }

  // ---- epilogue: bias + relu, store f16, per-channel sum/sumsq ----
  const int ch0 = lane & 15;
  const int ch1 = 16 + ch0;
  float ls0 = 0.f, lq0 = 0.f, ls1 = 0.f, lq1 = 0.f;
#pragma unroll
  for (int v = 0; v < 8; ++v) {
    int mm = v + ((lane >> 4) << 3);
    int pq = tile * 16 + mm;
    if (pq < NTOT) {
      int b2 = pq / (HOUT * WOUT);
      int r2 = pq % (HOUT * WOUT);
      int o2 = r2 / WOUT, o3 = r2 % WOUT;
      float val = fmaxf(acc0[v] + bias[ch0], 0.0f);
      y[((size_t)(b2 * 24 + ch0)) * (HOUT * WOUT) + o2 * WOUT + o3] = (_Float16)val;
      ls0 += val; lq0 += val * val;
      if (ch1 < 24) {
        float v1 = fmaxf(acc1[v] + bias[ch1], 0.0f);
        y[((size_t)(b2 * 24 + ch1)) * (HOUT * WOUT) + o2 * WOUT + o3] = (_Float16)v1;
        ls1 += v1; lq1 += v1 * v1;
      }
    }
  }
  ls0 += __shfl_xor(ls0, 16, 32);  lq0 += __shfl_xor(lq0, 16, 32);
  ls1 += __shfl_xor(ls1, 16, 32);  lq1 += __shfl_xor(lq1, 16, 32);
  if (lane < 16) { atomicAdd(&stats[ch0], ls0); atomicAdd(&stats[24 + ch0], lq0); }
  if (lane < 8)  { atomicAdd(&stats[ch1], ls1); atomicAdd(&stats[24 + ch1], lq1); }
}

// Finalize BN: scale = g/sqrt(var+eps), shift = b - mean*scale.
__global__ void bnfin_kernel(const float* __restrict__ stats,
                             const float* __restrict__ g,
                             const float* __restrict__ bb,
                             float n, float* __restrict__ aff)
{
  int c = threadIdx.x;
  if (c < 24) {
    float mean = stats[c] / n;
    float var  = stats[24 + c] / n - mean * mean;
    float sc   = g[c] * rsqrtf(var + 1e-5f);
    aff[c]      = sc;
    aff[24 + c] = bb[c] - mean * sc;
  }
}

// ---------------------------------------------------------------------------
// Recurrent / attention tail: one wave per batch element; all state stays
// distributed (one register per lane), broadcasts inline via __shfl.
// __launch_bounds__ min-waves hint caps VGPRs so the scheduler cannot batch
// every unrolled weight load at once (round-2 build hit 548 VGPRs).
// ---------------------------------------------------------------------------
template<int XD>
__device__ __forceinline__ float gru_step(float xd, float h,
    const float* __restrict__ wih, const float* __restrict__ whh,
    const float* __restrict__ bih, const float* __restrict__ bhh, int lane)
{
  float gir = bih[lane], giz = bih[32 + lane], gin = bih[64 + lane];
#pragma unroll
  for (int k = 0; k < XD; ++k) {
    float xv = __shfl(xd, k, 32);
    gir += xv * wih[lane * XD + k];
    giz += xv * wih[(32 + lane) * XD + k];
    gin += xv * wih[(64 + lane) * XD + k];
  }
  float ghr = bhh[lane], ghz = bhh[32 + lane], ghn = bhh[64 + lane];
#pragma unroll
  for (int k = 0; k < 32; ++k) {
    float hv = __shfl(h, k, 32);
    ghr += hv * whh[lane * 32 + k];
    ghz += hv * whh[(32 + lane) * 32 + k];
    ghn += hv * whh[(64 + lane) * 32 + k];
  }
  float r  = 1.0f / (1.0f + expf(-(gir + ghr)));
  float z  = 1.0f / (1.0f + expf(-(giz + ghz)));
  float nn = tanhf(gin + r * ghn);
  return (1.0f - z) * nn + z * h;
}

__global__ __launch_bounds__(128, 8) void recurrent_kernel(
    const _Float16* __restrict__ y4,  const float* __restrict__ aff4,
    const float* __restrict__ qst,    const float* __restrict__ gum,
    const float* __restrict__ k_blank,
    const float* __restrict__ rnn1_pad, const float* __restrict__ rnn1_start,
    const float* __restrict__ rnn2_h0,
    const float* __restrict__ g1_wih, const float* __restrict__ g1_whh,
    const float* __restrict__ g1_bih, const float* __restrict__ g1_bhh,
    const float* __restrict__ g2_wih, const float* __restrict__ g2_whh,
    const float* __restrict__ g2_bih, const float* __restrict__ g2_bhh,
    const float* __restrict__ q_w,    const float* __restrict__ q_b,
    const float* __restrict__ sq_pad, const float* __restrict__ sa_h0,
    const float* __restrict__ gq_wih, const float* __restrict__ gq_whh,
    const float* __restrict__ gq_bih, const float* __restrict__ gq_bhh,
    const float* __restrict__ ga_wih, const float* __restrict__ ga_whh,
    const float* __restrict__ ga_bih, const float* __restrict__ ga_bhh,
    const float* __restrict__ out_w,  const float* __restrict__ out_b,
    float* __restrict__ out)
{
  const int wave = (blockIdx.x * blockDim.x + threadIdx.x) >> 5;
  const int lane = threadIdx.x & 31;
  const int b = wave;
  const float NEG_INF = -__builtin_huge_valf();

  // Lane k<26 owns key k: ks[12] / vs[16] built from BN'd conv4 output.
  float ks[12], vs[16];
#pragma unroll
  for (int c = 0; c < 12; ++c) ks[c] = 0.0f;
#pragma unroll
  for (int v = 0; v < 16; ++v) vs[v] = 0.0f;
  if (lane < 25) {
    float cy = ((lane / 5) - 2) * 0.5f;
    float cx = ((lane % 5) - 2) * 0.5f;
#pragma unroll
    for (int c = 0; c < 10; ++c)
      ks[c] = (float)y4[(size_t)(b * 24 + c) * 25 + lane] * aff4[c] + aff4[24 + c];
    ks[10] = cy; ks[11] = cx;
#pragma unroll
    for (int v = 0; v < 14; ++v)
      vs[v] = (float)y4[(size_t)(b * 24 + 10 + v) * 25 + lane] * aff4[10 + v] + aff4[24 + 10 + v];
    vs[14] = cy; vs[15] = cx;
  } else if (lane == 25) {
#pragma unroll
    for (int c = 0; c < 12; ++c) ks[c] = k_blank[c];
  }

  float h1 = (lane < 11) ? qst[b * 11 + lane] : rnn1_pad[lane - 11];
  float h2 = rnn2_h0[lane];
  float inp_d = rnn1_start[lane & 15];   // distributed: lane v<16 owns inp_v
  float svA = 0.0f, svB = 0.0f;

#pragma unroll 1
  for (int t = 0; t < 2; ++t) {
    h1 = gru_step<16>(inp_d, h1, g1_wih, g1_whh, g1_bih, g1_bhh, lane);
    h2 = gru_step<32>(h1,    h2, g2_wih, g2_whh, g2_bih, g2_bhh, lane);

    // qs distributed: lane j<12 owns qs_j = q_b[j] + sum_k h2_k * q_w[j,k]
    int jq = (lane < 12) ? lane : 0;
    float qsd = q_b[jq];
#pragma unroll
    for (int k = 0; k < 32; ++k) qsd += __shfl(h2, k, 32) * q_w[jq * 32 + k];

    // sim for this lane's key + gumbel noise, log-softmax over 26 lanes
    float sim = 0.0f;
#pragma unroll
    for (int c = 0; c < 12; ++c) sim += ks[c] * __shfl(qsd, c, 32);
    float u = (lane < 26) ? gum[t * (BSZ * 26) + b * 26 + lane] : 0.5f;
    float noise = -logf(-logf(u + 1e-20f) + 1e-20f);
    float tt = (lane < 26) ? (sim + noise) * 2.0f : NEG_INF;
    float mx = tt;
#pragma unroll
    for (int s = 16; s >= 1; s >>= 1) mx = fmaxf(mx, __shfl_xor(mx, s, 32));
    float e = (lane < 26) ? expf(tt - mx) : 0.0f;
    float se = e;
#pragma unroll
    for (int s = 16; s >= 1; s >>= 1) se += __shfl_xor(se, s, 32);
    float wlog = tt - (mx + logf(se));
    float wk = (lane < 26) ? wlog : 0.0f;   // log-probs used directly, per ref

    // new inp (distributed): reduce one component at a time, keep only lane v
    float ninp = 0.0f;
#pragma unroll
    for (int v = 0; v < 16; ++v) {
      float cvv = wk * vs[v];
#pragma unroll
      for (int s = 16; s >= 1; s >>= 1) cvv += __shfl_xor(cvv, s, 32);
      if (lane == v) ninp = cvv;
    }
    inp_d = ninp;
    if (t == 0) svA = ninp; else svB = ninp;
  }

  float hq = (lane < 11) ? qst[b * 11 + lane] : sq_pad[lane - 11];
  float ha = sa_h0[lane];
#pragma unroll 1
  for (int t = 0; t < 2; ++t) {
    float xv = (t == 0) ? svA : svB;
    hq = gru_step<16>(xv, hq, gq_wih, gq_whh, gq_bih, gq_bhh, lane);
    ha = gru_step<32>(hq, ha, ga_wih, ga_whh, ga_bih, ga_bhh, lane);
  }

  // ans = ha @ out_w.T + out_b; log-softmax over 10 lanes (all inline shfl)
  int jo = (lane < 10) ? lane : 0;
  float ans = out_b[jo];
#pragma unroll
  for (int k = 0; k < 32; ++k) ans += __shfl(ha, k, 32) * out_w[jo * 32 + k];
  float tt = (lane < 10) ? ans : NEG_INF;
  float mx = tt;
#pragma unroll
  for (int s = 16; s >= 1; s >>= 1) mx = fmaxf(mx, __shfl_xor(mx, s, 32));
  float e = (lane < 10) ? expf(tt - mx) : 0.0f;
  float se = e;
#pragma unroll
  for (int s = 16; s >= 1; s >>= 1) se += __shfl_xor(se, s, 32);
  if (lane < 10) out[b * 10 + lane] = tt - (mx + logf(se));
}

// ---------------------------------------------------------------------------
extern "C" void kernel_launch(void* const* d_in, const int* in_sizes, int n_in,
                              void* d_out, int out_size, void* d_ws, size_t ws_size,
                              hipStream_t stream) {
  const float* img     = (const float*)d_in[0];
  const float* qst     = (const float*)d_in[1];
  const float* gum     = (const float*)d_in[2];
  const float* conv1_w = (const float*)d_in[3];
  const float* conv1_b = (const float*)d_in[4];
  const float* bn1_g   = (const float*)d_in[5];
  const float* bn1_b   = (const float*)d_in[6];
  const float* conv2_w = (const float*)d_in[7];
  const float* conv2_b = (const float*)d_in[8];
  const float* bn2_g   = (const float*)d_in[9];
  const float* bn2_b   = (const float*)d_in[10];
  const float* conv3_w = (const float*)d_in[11];
  const float* conv3_b = (const float*)d_in[12];
  const float* bn3_g   = (const float*)d_in[13];
  const float* bn3_b   = (const float*)d_in[14];
  const float* conv4_w = (const float*)d_in[15];
  const float* conv4_b = (const float*)d_in[16];
  const float* bn4_g   = (const float*)d_in[17];
  const float* bn4_b   = (const float*)d_in[18];
  const float* k_blank = (const float*)d_in[19];
  const float* rnn1_pad   = (const float*)d_in[20];
  const float* rnn1_start = (const float*)d_in[21];
  const float* rnn2_h0    = (const float*)d_in[22];
  const float* g1_wih = (const float*)d_in[23];
  const float* g1_whh = (const float*)d_in[24];
  const float* g1_bih = (const float*)d_in[25];
  const float* g1_bhh = (const float*)d_in[26];
  const float* g2_wih = (const float*)d_in[27];
  const float* g2_whh = (const float*)d_in[28];
  const float* g2_bih = (const float*)d_in[29];
  const float* g2_bhh = (const float*)d_in[30];
  const float* q_w    = (const float*)d_in[31];
  const float* q_b    = (const float*)d_in[32];
  const float* sq_pad = (const float*)d_in[33];
  const float* sa_h0  = (const float*)d_in[34];
  const float* gq_wih = (const float*)d_in[35];
  const float* gq_whh = (const float*)d_in[36];
  const float* gq_bih = (const float*)d_in[37];
  const float* gq_bhh = (const float*)d_in[38];
  const float* ga_wih = (const float*)d_in[39];
  const float* ga_whh = (const float*)d_in[40];
  const float* ga_bih = (const float*)d_in[41];
  const float* ga_bhh = (const float*)d_in[42];
  const float* out_w  = (const float*)d_in[43];
  const float* out_b  = (const float*)d_in[44];
  float* out = (float*)d_out;

  // ws layout (f16 activations; y3/y4 alias the dead y1 region)
  _Float16* y1 = (_Float16*)d_ws;                  // 1024*24*38*38
  _Float16* y2 = y1 + (size_t)1024 * 24 * 38 * 38; // 1024*24*19*19
  float* stats = (float*)(y2 + (size_t)1024 * 24 * 19 * 19); // 4*48
  float* aff   = stats + 192;                      // 4*48
  _Float16* y3 = y1;                               // 1024*24*10*10
  _Float16* y4 = y1 + (size_t)1024 * 24 * 10 * 10; // 1024*24*5*5

  hipMemsetAsync(stats, 0, 192 * sizeof(float), stream);

  { // conv1: [B,3,75,75] -> [B,24,38,38]
    int tiles = (1024 * 38 * 38 + 15) / 16, blocks = (tiles + 7) / 8;
    conv_wmma_kernel<float, 3, 75, 75, 38, 38><<<blocks, 256, 0, stream>>>(
        img, conv1_w, conv1_b, nullptr, y1, stats + 0);
    bnfin_kernel<<<1, 32, 0, stream>>>(stats + 0, bn1_g, bn1_b,
                                       (float)(1024 * 38 * 38), aff + 0);
  }
  { // conv2: -> [B,24,19,19]
    int tiles = (1024 * 19 * 19 + 15) / 16, blocks = (tiles + 7) / 8;
    conv_wmma_kernel<_Float16, 24, 38, 38, 19, 19><<<blocks, 256, 0, stream>>>(
        y1, conv2_w, conv2_b, aff + 0, y2, stats + 48);
    bnfin_kernel<<<1, 32, 0, stream>>>(stats + 48, bn2_g, bn2_b,
                                       (float)(1024 * 19 * 19), aff + 48);
  }
  { // conv3: -> [B,24,10,10]
    int tiles = (1024 * 10 * 10 + 15) / 16, blocks = (tiles + 7) / 8;
    conv_wmma_kernel<_Float16, 24, 19, 19, 10, 10><<<blocks, 256, 0, stream>>>(
        y2, conv3_w, conv3_b, aff + 48, y3, stats + 96);
    bnfin_kernel<<<1, 32, 0, stream>>>(stats + 96, bn3_g, bn3_b,
                                       (float)(1024 * 10 * 10), aff + 96);
  }
  { // conv4: -> [B,24,5,5]
    int tiles = (1024 * 5 * 5 + 15) / 16, blocks = (tiles + 7) / 8;
    conv_wmma_kernel<_Float16, 24, 10, 10, 5, 5><<<blocks, 256, 0, stream>>>(
        y3, conv4_w, conv4_b, aff + 96, y4, stats + 144);
    bnfin_kernel<<<1, 32, 0, stream>>>(stats + 144, bn4_g, bn4_b,
                                       (float)(1024 * 5 * 5), aff + 144);
  }
  // recurrent tail: 1024 waves (one per batch row), 4 waves per block
  recurrent_kernel<<<256, 128, 0, stream>>>(
      y4, aff + 144, qst, gum, k_blank, rnn1_pad, rnn1_start, rnn2_h0,
      g1_wih, g1_whh, g1_bih, g1_bhh, g2_wih, g2_whh, g2_bih, g2_bhh,
      q_w, q_b, sq_pad, sa_h0,
      gq_wih, gq_whh, gq_bih, gq_bhh, ga_wih, ga_whh, ga_bih, ga_bhh,
      out_w, out_b, out);
}
